// MultiHeadAttn_70600672411667
// MI455X (gfx1250) — matrix operand order repaired and checked by reading, not compiled, not run on hardware
//
#include <hip/hip_runtime.h>

// Problem constants
#define WS_B   4
#define WS_NQ  2048
#define WS_NK  2048
#define WS_D   1024
#define WS_H   8
#define WS_DH  128
#define WS_M   (WS_B * WS_NQ)   // 8192 rows for all row-major [row][D] tensors

typedef __attribute__((ext_vector_type(16))) __bf16 v16bf;
typedef __attribute__((ext_vector_type(8)))  __bf16 bf16x8;
typedef __attribute__((ext_vector_type(8)))  float  v8f;

#define NEGINF (-__builtin_inff())

__device__ __forceinline__ v8f zero8() {
  v8f z;
#pragma unroll
  for (int i = 0; i < 8; ++i) z[i] = 0.0f;
  return z;
}

__device__ __forceinline__ v16bf combine16(bf16x8 lo, bf16x8 hi) {
  return __builtin_shufflevector(lo, hi, 0, 1, 2, 3, 4, 5, 6, 7,
                                 8, 9, 10, 11, 12, 13, 14, 15);
}

__device__ __forceinline__ v8f wmma_bf16(v16bf a, v16bf b, v8f c) {
  // D = A(16x32) * B(32x16) + C(16x16), f32 accumulate
  return __builtin_amdgcn_wmma_f32_16x16x32_bf16(false, a, false, b, (short)0, c,
                                                 false, false);
}

// ---------------------------------------------------------------------------
// Fragment loaders per CDNA5 ISA VGPR layouts (wave32).
// A (16x32 MxK): lane&15 = M row; lane&16 -> K sub-half (+8);
//                lane holds K = kb+0..7 and kb+16..23 (two 16B runs).
// B (32x16 KxN): lane&15 = N col; lane&16 -> K sub-half (+8);
//                lane holds K = kb+0..7 and kb+16..23 (two 16B runs).
// B source must be stored row-major as [n][k] (transposed operand).
// ---------------------------------------------------------------------------
__device__ __forceinline__ v16bf afrag_bf16(const __bf16* base, int ld, int row0,
                                            int k0, int lane) {
  const __bf16* r = base + (size_t)(row0 + (lane & 15)) * ld + k0 + ((lane & 16) ? 8 : 0);
  bf16x8 lo = *(const bf16x8*)(r);        // K = kb+0..7
  bf16x8 hi = *(const bf16x8*)(r + 16);   // K = kb+16..23
  return combine16(lo, hi);
}

__device__ __forceinline__ v16bf bfrag_bf16(const __bf16* base, int ld, int n0,
                                            int k0, int lane) {
  const __bf16* r = base + (size_t)(n0 + (lane & 15)) * ld + k0 + ((lane & 16) ? 8 : 0);
  bf16x8 lo = *(const bf16x8*)(r);        // K = kb+0..7
  bf16x8 hi = *(const bf16x8*)(r + 16);   // K = kb+16..23
  return combine16(lo, hi);
}

// ---------------------------------------------------------------------------
// f32 -> bf16 bulk convert, 8 elements/thread, b128 in / b128 out
// ---------------------------------------------------------------------------
__global__ __launch_bounds__(256) void mha_cvt_bf16(const float* __restrict__ X,
                                                    __bf16* __restrict__ Y, int n) {
  int i = (blockIdx.x * 256 + threadIdx.x) * 8;
  if (i + 8 > n) return;
  float4 f0 = *(const float4*)(X + i);
  float4 f1 = *(const float4*)(X + i + 4);
  bf16x8 o;
  o[0] = (__bf16)f0.x; o[1] = (__bf16)f0.y; o[2] = (__bf16)f0.z; o[3] = (__bf16)f0.w;
  o[4] = (__bf16)f1.x; o[5] = (__bf16)f1.y; o[6] = (__bf16)f1.z; o[7] = (__bf16)f1.w;
  *(bf16x8*)(Y + i) = o;
}

// ---------------------------------------------------------------------------
// Projection GEMM: C[M,N] = A[M,K](bf16) * Wb[N,K](bf16)^T  (K=N=1024)
// Block = 8 waves (2Mx4N), 128x256 tile; wave = 64x64 (4x4 WMMA accs,
// 16 WMMA per 8 fragment loads).
// ---------------------------------------------------------------------------
__global__ __launch_bounds__(256) void mha_gemm_proj(const __bf16* __restrict__ A,
                                                     const __bf16* __restrict__ Wb,
                                                     float* __restrict__ outF,
                                                     __bf16* __restrict__ outB) {
  const int K = WS_D, N = WS_D;
  int lane = threadIdx.x & 31;
  int w = threadIdx.x >> 5;
  int m0 = blockIdx.y * 128 + (w & 1) * 64;
  int n0 = blockIdx.x * 256 + (w >> 1) * 64;

  v8f acc[4][4];
#pragma unroll
  for (int i = 0; i < 4; ++i)
#pragma unroll
    for (int j = 0; j < 4; ++j) acc[i][j] = zero8();

  for (int k0 = 0; k0 < K; k0 += 32) {
    v16bf af[4], bfr[4];
#pragma unroll
    for (int i = 0; i < 4; ++i) af[i] = afrag_bf16(A, K, m0 + 16 * i, k0, lane);
#pragma unroll
    for (int j = 0; j < 4; ++j) bfr[j] = bfrag_bf16(Wb, K, n0 + 16 * j, k0, lane);
#pragma unroll
    for (int i = 0; i < 4; ++i)
#pragma unroll
      for (int j = 0; j < 4; ++j) acc[i][j] = wmma_bf16(af[i], bfr[j], acc[i][j]);
  }

  int half8 = (lane & 16) ? 8 : 0;
  if (outF) {
#pragma unroll
    for (int i = 0; i < 4; ++i)
#pragma unroll
      for (int j = 0; j < 4; ++j)
#pragma unroll
        for (int r = 0; r < 8; ++r) {
          size_t m = (size_t)(m0 + 16 * i + r + half8);
          size_t n = (size_t)(n0 + 16 * j + (lane & 15));
          outF[m * N + n] = acc[i][j][r];
        }
  }
  if (outB) {
#pragma unroll
    for (int i = 0; i < 4; ++i)
#pragma unroll
      for (int j = 0; j < 4; ++j)
#pragma unroll
        for (int r = 0; r < 8; ++r) {
          size_t m = (size_t)(m0 + 16 * i + r + half8);
          size_t n = (size_t)(n0 + 16 * j + (lane & 15));
          outB[m * N + n] = (__bf16)acc[i][j][r];
        }
  }
}

// ---------------------------------------------------------------------------
// Output GEMM: Y = relu(A[M,K](bf16) * Wb[N,K](bf16)^T + bias) + resid
// ---------------------------------------------------------------------------
__global__ __launch_bounds__(256) void mha_gemm_out(const __bf16* __restrict__ A,
                                                    const __bf16* __restrict__ Wb,
                                                    const float* __restrict__ bias,
                                                    const float* __restrict__ resid,
                                                    float* __restrict__ Y) {
  const int K = WS_D, N = WS_D;
  int lane = threadIdx.x & 31;
  int w = threadIdx.x >> 5;
  int m0 = blockIdx.y * 128 + (w & 1) * 64;
  int n0 = blockIdx.x * 256 + (w >> 1) * 64;

  v8f acc[4][4];
#pragma unroll
  for (int i = 0; i < 4; ++i)
#pragma unroll
    for (int j = 0; j < 4; ++j) acc[i][j] = zero8();

  for (int k0 = 0; k0 < K; k0 += 32) {
    v16bf af[4], bfr[4];
#pragma unroll
    for (int i = 0; i < 4; ++i) af[i] = afrag_bf16(A, K, m0 + 16 * i, k0, lane);
#pragma unroll
    for (int j = 0; j < 4; ++j) bfr[j] = bfrag_bf16(Wb, K, n0 + 16 * j, k0, lane);
#pragma unroll
    for (int i = 0; i < 4; ++i)
#pragma unroll
      for (int j = 0; j < 4; ++j) acc[i][j] = wmma_bf16(af[i], bfr[j], acc[i][j]);
  }

  int half8 = (lane & 16) ? 8 : 0;
#pragma unroll
  for (int i = 0; i < 4; ++i)
#pragma unroll
    for (int j = 0; j < 4; ++j)
#pragma unroll
      for (int r = 0; r < 8; ++r) {
        size_t m = (size_t)(m0 + 16 * i + r + half8);
        size_t n = (size_t)(n0 + 16 * j + (lane & 15));
        float vv = acc[i][j][r] + bias[n];
        vv = fmaxf(vv, 0.0f) + resid[m * N + n];
        Y[m * N + n] = vv;
      }
}

// ---------------------------------------------------------------------------
// Flash attention: block = 4 waves, 64 q-rows (16 per wave), one (b,h) pair.
// K tile staged via global_load_async_to_lds_b128 (ASYNCcnt), V tile staged
// transposed via regular LDS stores, S and P*V via bf16 WMMA.
// ---------------------------------------------------------------------------
__global__ __launch_bounds__(128) void mha_attn(const __bf16* __restrict__ qp,
                                                const __bf16* __restrict__ kp,
                                                const __bf16* __restrict__ vp,
                                                const unsigned char* __restrict__ mask,
                                                float* __restrict__ attOut) {
  __shared__ __bf16 Ktile[32][136];     // K tile [kk][d], ld=136 (16B-aligned rows, conflict-free)
  __shared__ __bf16 Vt[WS_DH][40];      // V^T tile [d][kk], ld=40 (16B-aligned rows)
  __shared__ __bf16 Pst[4][16][32];     // per-wave P staging

  int lane = threadIdx.x & 31;
  int w = threadIdx.x >> 5;
  int bh = blockIdx.y;
  int b = bh / WS_H, h = bh % WS_H;
  int q0 = blockIdx.x * 64 + w * 16;
  int qrow = b * WS_NQ + q0;
  int dbase = h * WS_DH;
  const unsigned char* mk = mask + (size_t)b * WS_NK;

  // Q fragments for this wave's 16 rows; K-dim = Dh = 128 -> 4 A-fragments
  v16bf qf[4];
#pragma unroll
  for (int dk = 0; dk < 4; ++dk)
    qf[dk] = afrag_bf16(qp, WS_D, qrow, dbase + 32 * dk, lane);

  v8f o[8];
#pragma unroll
  for (int dt = 0; dt < 8; ++dt) o[dt] = zero8();
  float mrow[8], lrow[8];
#pragma unroll
  for (int r = 0; r < 8; ++r) { mrow[r] = NEGINF; lrow[r] = 0.0f; }

  int half8 = (lane & 16) ? 8 : 0;

  for (int kc = 0; kc < WS_NK; kc += 32) {
    __syncthreads();

    // --- Async stage K tile: 32x128 bf16 = 512 x 16B chunks, 4 per thread ---
    {
      const __bf16* ksrc = kp + (size_t)(b * WS_NK + kc) * WS_D + dbase;
#pragma unroll
      for (int i = 0; i < 4; ++i) {
        int idx = threadIdx.x + i * 128;   // 0..511
        int row = idx >> 4;                // key 0..31
        int jj  = idx & 15;                // 16B chunk within row
        unsigned dst = (unsigned)(uintptr_t)(&Ktile[row][jj * 8]);
        const __bf16* src = ksrc + (size_t)row * WS_D + jj * 8;
        asm volatile("global_load_async_to_lds_b128 %0, %1, off"
                     :: "v"(dst), "v"(src) : "memory");
      }
    }

    // --- Stage V^T tile: coalesced read along d, scatter into LDS ---
#pragma unroll 4
    for (int i = 0; i < 32; ++i) {
      int e = threadIdx.x + i * 128;
      int c = e >> 7;            // key index within chunk 0..31
      int d = e & 127;           // head-dim 0..127
      Vt[d][c] = vp[(size_t)(b * WS_NK + kc + c) * WS_D + dbase + d];
    }

    asm volatile("s_wait_asynccnt 0" ::: "memory");
    __syncthreads();

    // S = Q * K^T : two 16x16 tiles (columns kc..kc+15, kc+16..kc+31)
    v8f s0 = zero8(), s1 = zero8();
#pragma unroll
    for (int dk = 0; dk < 4; ++dk) {
      v16bf kf0 = bfrag_bf16(&Ktile[0][0], 136, 0,  dk * 32, lane);
      v16bf kf1 = bfrag_bf16(&Ktile[0][0], 136, 16, dk * 32, lane);
      s0 = wmma_bf16(qf[dk], kf0, s0);
      s1 = wmma_bf16(qf[dk], kf1, s1);
    }

    bool msk0 = mk[kc + (lane & 15)] != 0;
    bool msk1 = mk[kc + 16 + (lane & 15)] != 0;

#pragma unroll
    for (int r = 0; r < 8; ++r) {
      float v0 = msk0 ? NEGINF : s0[r] * 0.03125f;   // 1/sqrt(1024)
      float v1 = msk1 ? NEGINF : s1[r] * 0.03125f;
      float rm = fmaxf(v0, v1);
      rm = fmaxf(rm, __shfl_xor(rm, 1, 32));
      rm = fmaxf(rm, __shfl_xor(rm, 2, 32));
      rm = fmaxf(rm, __shfl_xor(rm, 4, 32));
      rm = fmaxf(rm, __shfl_xor(rm, 8, 32));
      float mold = mrow[r];
      float mnew = fmaxf(mold, rm);
      bool allinf = (mnew == NEGINF);
      float alpha = allinf ? 1.0f : __expf(mold - mnew);
      float p0 = allinf ? 0.0f : __expf(v0 - mnew);
      float p1 = allinf ? 0.0f : __expf(v1 - mnew);
      float ps = p0 + p1;
      ps += __shfl_xor(ps, 1, 32);
      ps += __shfl_xor(ps, 2, 32);
      ps += __shfl_xor(ps, 4, 32);
      ps += __shfl_xor(ps, 8, 32);
      lrow[r] = lrow[r] * alpha + ps;
      mrow[r] = mnew;
#pragma unroll
      for (int dt = 0; dt < 8; ++dt) o[dt][r] = o[dt][r] * alpha;
      Pst[w][r + half8][lane & 15]        = (__bf16)p0;
      Pst[w][r + half8][16 + (lane & 15)] = (__bf16)p1;
    }

    // Reload P in A-fragment layout (same-wave LDS dependency, in-order DS)
    {
      const __bf16* pr = &Pst[w][lane & 15][(lane & 16) ? 8 : 0];
      bf16x8 lo = *(const bf16x8*)(pr);
      bf16x8 hi = *(const bf16x8*)(pr + 16);
      v16bf pf = combine16(lo, hi);

      // O += P * V  (B-fragments from transposed V tile in LDS)
#pragma unroll
      for (int dt = 0; dt < 8; ++dt) {
        v16bf vf = bfrag_bf16(&Vt[0][0], 40, dt * 16, 0, lane);
        o[dt] = wmma_bf16(pf, vf, o[dt]);
      }
    }
  }

  // Normalize and write out (fully-masked rows -> 0, matching NaN->0 ref)
#pragma unroll
  for (int dt = 0; dt < 8; ++dt)
#pragma unroll
    for (int r = 0; r < 8; ++r) {
      float l = lrow[r];
      float vv = (l > 0.0f) ? o[dt][r] / l : 0.0f;
      size_t row = (size_t)(b * WS_NQ + q0 + r + half8);
      attOut[row * WS_D + dbase + dt * 16 + (lane & 15)] = vv;
    }
}

// ---------------------------------------------------------------------------
// LayerNorm over D=1024: y = LN(X + R) * g + b;  R optional
// ---------------------------------------------------------------------------
__global__ __launch_bounds__(256) void mha_ln(const float* __restrict__ X,
                                              const float* __restrict__ R,
                                              const float* __restrict__ g,
                                              const float* __restrict__ bb,
                                              float* __restrict__ outF,
                                              __bf16* __restrict__ outB) {
  __shared__ float red[256];
  int row = blockIdx.x;
  int t = threadIdx.x;
  const float* x = X + (size_t)row * WS_D;
  const float* r = R ? (R + (size_t)row * WS_D) : nullptr;

  float vals[4];
  float s = 0.0f;
#pragma unroll
  for (int i = 0; i < 4; ++i) {
    int c = t + i * 256;
    float v = x[c];
    if (r) v += r[c];
    vals[i] = v;
    s += v;
  }
  red[t] = s;
  __syncthreads();
  for (int off = 128; off > 0; off >>= 1) {
    if (t < off) red[t] += red[t + off];
    __syncthreads();
  }
  float mu = red[0] * (1.0f / WS_D);
  __syncthreads();

  float s2 = 0.0f;
#pragma unroll
  for (int i = 0; i < 4; ++i) {
    float d = vals[i] - mu;
    s2 += d * d;
  }
  red[t] = s2;
  __syncthreads();
  for (int off = 128; off > 0; off >>= 1) {
    if (t < off) red[t] += red[t + off];
    __syncthreads();
  }
  float inv = rsqrtf(red[0] * (1.0f / WS_D) + 1e-5f);

#pragma unroll
  for (int i = 0; i < 4; ++i) {
    int c = t + i * 256;
    float y = (vals[i] - mu) * inv * g[c] + bb[c];
    if (outF) outF[(size_t)row * WS_D + c] = y;
    if (outB) outB[(size_t)row * WS_D + c] = (__bf16)y;
  }
}

// ---------------------------------------------------------------------------
extern "C" void kernel_launch(void* const* d_in, const int* in_sizes, int n_in,
                              void* d_out, int out_size, void* d_ws, size_t ws_size,
                              hipStream_t stream) {
  (void)in_sizes; (void)n_in; (void)out_size; (void)ws_size;

  const float* q    = (const float*)d_in[0];
  const float* k    = (const float*)d_in[1];
  const float* v    = (const float*)d_in[2];
  const unsigned char* mask = (const unsigned char*)d_in[3];
  const float* Wq   = (const float*)d_in[4];
  const float* Wk   = (const float*)d_in[5];
  const float* Wv   = (const float*)d_in[6];
  const float* Wout = (const float*)d_in[7];
  const float* bout = (const float*)d_in[8];
  const float* g1   = (const float*)d_in[9];
  const float* b1   = (const float*)d_in[10];
  const float* g2   = (const float*)d_in[11];
  const float* b2   = (const float*)d_in[12];
  float* out = (float*)d_out;

  // Workspace layout (bytes); bf16 input copies alias later-written buffers
  // (dead after projections -> safe to overwrite).
  char* ws = (char*)d_ws;
  __bf16* WqB   = (__bf16*)(ws + 0);                       //  2097152
  __bf16* WkB   = (__bf16*)(ws + (size_t)2097152);         //  2097152
  __bf16* WvB   = (__bf16*)(ws + (size_t)4194304);         //  2097152
  __bf16* WoB   = (__bf16*)(ws + (size_t)6291456);         //  2097152
  float*  qpF   = (float*)(ws + (size_t)8388608);          // 33554432  qp f32
  __bf16* qpB   = (__bf16*)(ws + (size_t)41943040);        // 16777216  qp bf16
  __bf16* kpB   = (__bf16*)(ws + (size_t)58720256);        // 16777216  kp bf16
  __bf16* vpB   = (__bf16*)(ws + (size_t)75497472);        // 16777216  vp bf16
  float*  attF  = (float*)(ws + (size_t)92274688);         // 33554432  att / y tmp
  float*  out1F = (float*)(ws + (size_t)125829120);        // 33554432  LN1 out f32
  __bf16* out1B = (__bf16*)(ws + (size_t)159383552);       // 16777216  LN1 out bf16
  // Aliases (consumed before their hosts are written):
  __bf16* qB = (__bf16*)(ws + (size_t)92274688);           // aliases attF
  __bf16* kB = (__bf16*)(ws + (size_t)125829120);          // aliases out1F
  __bf16* vB = (__bf16*)(ws + (size_t)159383552);          // aliases out1B

  const int wElems  = WS_D * WS_D;                         // 1048576
  const int inElems = WS_M * WS_D;                         // 8388608
  int cvtGridW  = (wElems / 8 + 255) / 256;                // 512
  int cvtGridIn = (inElems / 8 + 255) / 256;               // 4096

  dim3 gemmGrid(WS_D / 256, WS_M / 128);   // (4, 64)
  dim3 attnGrid(WS_NQ / 64, WS_B * WS_H);  // (32, 32)

  // Pre-convert weights + activations to bf16 (pure-bf16 GEMM inner loops)
  mha_cvt_bf16<<<cvtGridW, 256, 0, stream>>>(Wq, WqB, wElems);
  mha_cvt_bf16<<<cvtGridW, 256, 0, stream>>>(Wk, WkB, wElems);
  mha_cvt_bf16<<<cvtGridW, 256, 0, stream>>>(Wv, WvB, wElems);
  mha_cvt_bf16<<<cvtGridW, 256, 0, stream>>>(Wout, WoB, wElems);
  mha_cvt_bf16<<<cvtGridIn, 256, 0, stream>>>(q, qB, inElems);
  mha_cvt_bf16<<<cvtGridIn, 256, 0, stream>>>(k, kB, inElems);
  mha_cvt_bf16<<<cvtGridIn, 256, 0, stream>>>(v, vB, inElems);

  // Projections (qB/kB/vB consumed here; their regions rewritten later)
  mha_gemm_proj<<<gemmGrid, 256, 0, stream>>>(qB, WqB, qpF, qpB);
  mha_gemm_proj<<<gemmGrid, 256, 0, stream>>>(kB, WkB, nullptr, kpB);
  mha_gemm_proj<<<gemmGrid, 256, 0, stream>>>(vB, WvB, nullptr, vpB);

  // Attention (overwrites qB region with attF -- qB is dead)
  mha_attn<<<attnGrid, 128, 0, stream>>>(qpB, kpB, vpB, mask, attF);

  // LN1: LN(qp + att)  (overwrites kB/vB regions -- dead)
  mha_ln<<<WS_M, 256, 0, stream>>>(qpF, attF, g1, b1, out1F, out1B);

  // out GEMM: relu(out1 * Wout^T + bout) + out1  -> attF (reused as tmp)
  mha_gemm_out<<<gemmGrid, 256, 0, stream>>>(out1B, WoB, bout, out1F, attF);

  // LN2 -> final output
  mha_ln<<<WS_M, 256, 0, stream>>>(attF, nullptr, g2, b2, out, nullptr);
}
